// single_att_head_4741643895723
// MI455X (gfx1250) — compile-verified, matrix operand order checked by reading, not run on hardware
//
#include <hip/hip_runtime.h>
#include <math.h>

// ---------------------------------------------------------------------------
// Causal single-attention-head for MI455X (gfx1250, wave32).
//   0) Wb = bf16(W); xbT[b][t][j] = bf16(x[b][j][t])          (pre-pass)
//   1) xwT[b][t][i] = (W @ x[b])^T      bf16; per-lane ASYNC global->LDS
//   2) corr[b][t][s] = xwT xwT^T /sqrt(d), f32; TDM tensor_load_to_lds tiles
//   3) attnT[b][s][t] = causal softmax over t  (bf16, reuses xbT region)
//   4) out[b] = x[b] @ attn[b]          f32; sync A staging + async B
// All GEMMs: v_wmma_f32_16x16x32_bf16, 128x128x64 double-buffered LDS tiles.
// Workspace: 8 + 64 + 64 + 128 = 264 MB.
// ---------------------------------------------------------------------------

typedef __bf16 bf16;
typedef __attribute__((ext_vector_type(16))) __bf16 v16bf;
typedef __attribute__((ext_vector_type(8)))  __bf16 v8bf;
typedef __attribute__((ext_vector_type(4)))  __bf16 v4bf;
typedef __attribute__((ext_vector_type(8)))  float  v8f;
typedef __attribute__((ext_vector_type(4)))  float  v4f;
typedef __attribute__((ext_vector_type(4)))  unsigned int v4u;
typedef __attribute__((ext_vector_type(8)))  unsigned int v8u;

#define DIM    2048
#define NBATCH 8
#define BM 128
#define BN 128
#define BK 64          // two 32-deep WMMA K-halves per staged tile
#define TILE_B (BM * BK * 2)  // bytes per bf16 tile (16 KB)

// GEMM staging variants
#define VAR_ASYNC 0    // A,B: per-lane global_load_async_to_lds_b128
#define VAR_TDM   1    // A,B: TDM tensor_load_to_lds (wave 0 issues)
#define VAR_MIXED 2    // A: sync f32->bf16 ds stores; B: async

// ---- CDNA5 data movers -----------------------------------------------------
__device__ __forceinline__ uint32_t lds_off_of(const void* p) {
  return (uint32_t)(uintptr_t)p;  // LDS aperture: low 32 bits = LDS offset
}
__device__ __forceinline__ void async_b128(uint32_t lds_off, const void* g) {
  asm volatile("global_load_async_to_lds_b128 %0, %1, off"
               :: "v"(lds_off), "v"(g) : "memory");
}
template <int N> __device__ __forceinline__ void wait_async() {
  if constexpr (N == 8)      asm volatile("s_wait_asynccnt 0x8" ::: "memory");
  else if constexpr (N == 4) asm volatile("s_wait_asynccnt 0x4" ::: "memory");
  else                       asm volatile("s_wait_asynccnt 0x0" ::: "memory");
}

// TDM: one 2D tile (BM rows x BK cols, bf16, row stride DIM) global -> LDS.
__device__ __forceinline__ void tdm_load_tile(uint32_t lds_addr, const bf16* gtile) {
  const uint64_t ga = (uint64_t)(uintptr_t)gtile;
  v4u g0;
  g0[0] = 1u;                                   // count=1, user descriptor
  g0[1] = lds_addr;                             // LDS byte address
  g0[2] = (uint32_t)ga;                         // global_addr[31:0]
  g0[3] = (uint32_t)((ga >> 32) & 0x01FFFFFFu)  // global_addr[56:32]
          | (2u << 30);                         // type=2 ("image")
  v8u g1;
  g1[0] = 0x00010000u;                          // wg_mask=0, data_size=2B
  g1[1] = ((uint32_t)DIM & 0xFFFFu) << 16;      // tensor_dim0[15:0]
  g1[2] = ((uint32_t)DIM >> 16)                 // tensor_dim0[31:16]
          | (((uint32_t)DIM & 0xFFFFu) << 16);  // tensor_dim1[15:0]
  g1[3] = ((uint32_t)DIM >> 16)                 // tensor_dim1[31:16]
          | ((uint32_t)BK << 16);               // tile_dim0 = 64 elems
  g1[4] = (uint32_t)BM;                         // tile_dim1 = 128 rows
  g1[5] = (uint32_t)DIM;                        // tensor_dim0_stride[31:0]
  g1[6] = 0u;                                   // stride hi, dim1_stride lo
  g1[7] = 0u;
  asm volatile("tensor_load_to_lds %0, %1" :: "s"(g0), "s"(g1) : "memory");
}

// ---- per-lane staging helpers (256 threads) -------------------------------
// bf16 tile rows -> LDS, 16B per async op, 4 ops/thread.
__device__ __forceinline__ void stage_async_tile(const bf16* G, int row0, int col0,
                                                 uint32_t lds_base, int tid) {
  const int r   = tid >> 3;        // 0..31
  const int seg = (tid & 7) * 8;   // bf16 elems 0..56
#pragma unroll
  for (int rr = 0; rr < BM; rr += 32) {
    const bf16* g = G + (size_t)(row0 + r + rr) * DIM + col0 + seg;
    async_b128(lds_base + (uint32_t)(((r + rr) * BK + seg) * 2), g);
  }
}

// f32 -> bf16 sync staging: As[m][k] = A[(m0+m)*DIM + k0+k]
__device__ __forceinline__ void stage_A_f32(const float* A, int m0, int k0,
                                            bf16* As, int tid) {
  const int row = tid >> 4;        // 0..15
  const int c4  = (tid & 15) * 4;  // 0..60
#pragma unroll
  for (int rr = 0; rr < BM; rr += 16) {
    v4f v = *(const v4f*)(A + (size_t)(m0 + row + rr) * DIM + k0 + c4);
    v4bf o;
    o[0] = (bf16)v[0]; o[1] = (bf16)v[1]; o[2] = (bf16)v[2]; o[3] = (bf16)v[3];
    *(v4bf*)(As + (row + rr) * BK + c4) = o;
  }
}

// 16x32 bf16 fragment from an LDS tile (row stride BK), K-half kh (0 or 32).
// lane L -> row = L%16, half = L/16; K runs [kh+8h, +8) and [kh+16+8h, +8).
__device__ __forceinline__ v16bf load_frag(const bf16* base, int row0, int kh,
                                           int lane) {
  const int r    = row0 + (lane & 15);
  const int half = (lane >> 4) & 1;
  const bf16* p  = base + r * BK + kh + half * 8;
  v8bf lo = *(const v8bf*)(p);
  v8bf hi = *(const v8bf*)(p + 16);
  v16bf f;
#pragma unroll
  for (int i = 0; i < 8; ++i) { f[i] = lo[i]; f[8 + i] = hi[i]; }
  return f;
}

// ---- 128x128x64 double-buffered bf16-WMMA GEMM ----------------------------
// C[m][n] = scale * sum_k A[m][k] * B[k][n]; B[k][n] = Bg[n][k] (rows hold K).
template <int VAR, bool CAUSAL_SKIP, bool OUT_TRANS_BF16>
__global__ __launch_bounds__(256) void gemm_wmma(
    const void* __restrict__ Ag, const bf16* __restrict__ Bg,
    void* __restrict__ Cg, float scale,
    size_t strideA, size_t strideB, size_t strideC) {
  const int bn = blockIdx.x;
  const int bm = blockIdx.y;
  if (CAUSAL_SKIP && bm > bn) return;  // tile entirely in masked region (t > s)

  const int tid  = threadIdx.x;
  const int lane = tid & 31;
  const int wave = tid >> 5;
  const int wm   = (wave & 3) * 32;   // 4 waves tile M
  const int wn   = (wave >> 2) * 64;  // 2 waves tile N

  __shared__ __align__(16) bf16 As[2][BM * BK];
  __shared__ __align__(16) bf16 Bs[2][BN * BK];
  const uint32_t As_off = lds_off_of(As);
  const uint32_t Bs_off = lds_off_of(Bs);

  const bf16*  Ab = (const bf16*)Ag + (size_t)blockIdx.z * strideA;
  const float* Af = (const float*)Ag + (size_t)blockIdx.z * strideA;
  const bf16*  Bb = Bg + (size_t)blockIdx.z * strideB;

  auto issue = [&](int buf, int k0) {
    if constexpr (VAR == VAR_TDM) {
      if (wave == 0) {
        tdm_load_tile(As_off + buf * TILE_B,
                      Ab + (size_t)(bm * BM) * DIM + k0);
        tdm_load_tile(Bs_off + buf * TILE_B,
                      Bb + (size_t)(bn * BN) * DIM + k0);
      }
    } else {
      if constexpr (VAR == VAR_ASYNC)
        stage_async_tile(Ab, bm * BM, k0, As_off + buf * TILE_B, tid);
      else {
        stage_A_f32(Af, bm * BM, k0, &As[buf][0], tid);
        if (k0 + BK < DIM)  // global_prefetch_b8 of upcoming A line
          __builtin_prefetch(Af + (size_t)(bm * BM + (tid >> 4)) * DIM + k0 + BK,
                             0, 1);
      }
      stage_async_tile(Bb, bn * BN, k0, Bs_off + buf * TILE_B, tid);
    }
  };
  auto wait_stage = [&](bool more_inflight) {
    if constexpr (VAR == VAR_TDM) {
      if (wave == 0) {
        if (more_inflight) __builtin_amdgcn_s_wait_tensorcnt(2);
        else               __builtin_amdgcn_s_wait_tensorcnt(0);
      }
    } else if constexpr (VAR == VAR_ASYNC) {
      if (more_inflight) wait_async<8>(); else wait_async<0>();
    } else {
      if (more_inflight) wait_async<4>(); else wait_async<0>();
    }
  };

  v8f zero = {0.f, 0.f, 0.f, 0.f, 0.f, 0.f, 0.f, 0.f};
  v8f acc[2][4];
#pragma unroll
  for (int mi = 0; mi < 2; ++mi)
#pragma unroll
    for (int ni = 0; ni < 4; ++ni) acc[mi][ni] = zero;

  issue(0, 0);  // prologue: first tile in flight

  for (int k0 = 0; k0 < DIM; k0 += BK) {
    const int cur = (k0 / BK) & 1;
    const bool have_next = (k0 + BK) < DIM;
    if (have_next) issue(cur ^ 1, k0 + BK);  // overlap copy with compute
    wait_stage(have_next);
    __syncthreads();  // current buffer resident for all waves

    const bf16* Abuf = &As[cur][0];
    const bf16* Bbuf = &Bs[cur][0];
#pragma unroll
    for (int kh = 0; kh < BK; kh += 32) {
      v16bf af[2], bfr[4];
#pragma unroll
      for (int mi = 0; mi < 2; ++mi)
        af[mi] = load_frag(Abuf, wm + mi * 16, kh, lane);
#pragma unroll
      for (int ni = 0; ni < 4; ++ni)
        bfr[ni] = load_frag(Bbuf, wn + ni * 16, kh, lane);
#pragma unroll
      for (int mi = 0; mi < 2; ++mi)
#pragma unroll
        for (int ni = 0; ni < 4; ++ni)
          acc[mi][ni] = __builtin_amdgcn_wmma_f32_16x16x32_bf16(
              false, af[mi], false, bfr[ni], (short)0, acc[mi][ni], false, false);
    }
    __syncthreads();  // reads done before buffer is overwritten
  }

  // Epilogue. D layout: lane -> N = lane%16, half = lane/16; vgpr r -> M = r+8h.
  const int half = lane >> 4;
  const int col  = lane & 15;
#pragma unroll
  for (int mi = 0; mi < 2; ++mi) {
#pragma unroll
    for (int ni = 0; ni < 4; ++ni) {
      const size_t gn  = (size_t)bn * BN + wn + ni * 16 + col;
      const size_t gm0 = (size_t)bm * BM + wm + mi * 16 + half * 8;
      if constexpr (OUT_TRANS_BF16) {
        v8bf o;
#pragma unroll
        for (int r = 0; r < 8; ++r) o[r] = (bf16)(acc[mi][ni][r] * scale);
        *(v8bf*)((bf16*)Cg + (size_t)blockIdx.z * strideC + gn * DIM + gm0) = o;
      } else {
#pragma unroll
        for (int r = 0; r < 8; ++r)
          ((float*)Cg)[(size_t)blockIdx.z * strideC + (gm0 + r) * DIM + gn] =
              acc[mi][ni][r] * scale;
      }
    }
  }
}

// ---- pre-pass: straight f32 -> bf16 convert -------------------------------
__global__ __launch_bounds__(256) void convert_bf16(const float* __restrict__ X,
                                                    bf16* __restrict__ Y) {
  const size_t i = ((size_t)blockIdx.x * 256 + threadIdx.x) * 8;
  v4f a = *(const v4f*)(X + i);
  v4f b = *(const v4f*)(X + i + 4);
  v8bf o;
#pragma unroll
  for (int j = 0; j < 4; ++j) { o[j] = (bf16)a[j]; o[4 + j] = (bf16)b[j]; }
  *(v8bf*)(Y + i) = o;
}

// ---- pre-pass: transposed f32 -> bf16: XT[t][j] = X[j][t] -----------------
__global__ __launch_bounds__(256) void transpose_bf16(const float* __restrict__ X,
                                                      bf16* __restrict__ XT) {
  __shared__ bf16 tile[64][72];
  const float* src = X + (size_t)blockIdx.z * DIM * DIM;
  bf16* dst = XT + (size_t)blockIdx.z * DIM * DIM;
  const int j0 = blockIdx.y * 64, t0 = blockIdx.x * 64;
  const int tid = threadIdx.x;

  const int rt = tid >> 4, ct = (tid & 15) * 4;
#pragma unroll
  for (int rr = 0; rr < 64; rr += 16) {
    v4f v = *(const v4f*)(src + (size_t)(j0 + rt + rr) * DIM + t0 + ct);
#pragma unroll
    for (int i = 0; i < 4; ++i) tile[ct + i][rt + rr] = (bf16)v[i];
  }
  __syncthreads();

  const int wr = tid >> 3, wc = (tid & 7) * 8;
#pragma unroll
  for (int rr = 0; rr < 64; rr += 32) {
    v8bf o;
#pragma unroll
    for (int i = 0; i < 8; ++i) o[i] = tile[wr + rr][wc + i];
    *(v8bf*)(dst + (size_t)(t0 + wr + rr) * DIM + j0 + wc) = o;
  }
}

// ---- causal column softmax over t; writes transposed bf16 attnT[s][t] -----
__global__ __launch_bounds__(256) void causal_softmax_T(const float* __restrict__ corr,
                                                        bf16* __restrict__ attnT) {
  const int s = blockIdx.x * 256 + threadIdx.x;
  const float* col = corr + (size_t)blockIdx.y * DIM * DIM + s;  // (t,s)
  bf16* row = attnT + (size_t)blockIdx.y * DIM * DIM + (size_t)s * DIM;

  float m = -INFINITY;
  for (int t = 0; t <= s; ++t) m = fmaxf(m, col[(size_t)t * DIM]);
  float sum = 0.f;
  for (int t = 0; t <= s; ++t) sum += __expf(col[(size_t)t * DIM] - m);
  const float inv = 1.0f / sum;
  for (int t = 0; t <= s; ++t)
    row[t] = (bf16)(__expf(col[(size_t)t * DIM] - m) * inv);
  for (int t = s + 1; t < DIM; ++t) row[t] = (bf16)0.0f;
}

// ---------------------------------------------------------------------------
extern "C" void kernel_launch(void* const* d_in, const int* in_sizes, int n_in,
                              void* d_out, int out_size, void* d_ws, size_t ws_size,
                              hipStream_t stream) {
  const float* x = (const float*)d_in[0];  // (B, D, T) f32
  const float* W = (const float*)d_in[1];  // (D, D)    f32
  float* out = (float*)d_out;              // (B, D, T) f32

  const size_t mat = (size_t)DIM * DIM;
  bf16*  Wb   = (bf16*)d_ws;                           //   8 MB
  bf16*  xbT  = Wb + mat;                              //  64 MB (reused as attnT)
  bf16*  xwT  = xbT + (size_t)NBATCH * mat;            //  64 MB
  float* corr = (float*)(xwT + (size_t)NBATCH * mat);  // 128 MB

  dim3 block(256);
  dim3 grid(DIM / BN, DIM / BM, NBATCH);
  const float inv_sqrt_d = 1.0f / sqrtf((float)DIM);

  // 0) one-time conversions / transpose
  convert_bf16<<<dim3((unsigned)(mat / (256 * 8))), block, 0, stream>>>(W, Wb);
  transpose_bf16<<<dim3(DIM / 64, DIM / 64, NBATCH), block, 0, stream>>>(x, xbT);

  // 1) xwT[b][t][i] = (W @ x[b])^T   (per-lane async staging; A=Wb shared)
  gemm_wmma<VAR_ASYNC, false, true>
      <<<grid, block, 0, stream>>>(Wb, xbT, xwT, 1.0f, 0, mat, mat);

  // 2) corr[b][t][s] = xw^T xw / sqrt(d)   (TDM staging; skip masked tiles)
  gemm_wmma<VAR_TDM, true, false>
      <<<grid, block, 0, stream>>>(xwT, xwT, corr, inv_sqrt_d, mat, mat, mat);

  // 3) attnT[b][s][t] = causal softmax over t   (reuses xbT storage)
  causal_softmax_T<<<dim3(DIM / 256, NBATCH), block, 0, stream>>>(corr, xbT);

  // 4) out[b] = x[b] @ attn[b]   (sync f32 A staging + async B)
  gemm_wmma<VAR_MIXED, false, false>
      <<<grid, block, 0, stream>>>(x, xbT, out, 1.0f, mat, mat, mat);
}
